// Mamba3Block_83537113907707
// MI455X (gfx1250) — compile-verified, matrix operand order checked by reading, not run on hardware
//
#include <hip/hip_runtime.h>
#include <hip/hip_bf16.h>
#include <math.h>

// ---------------------------------------------------------------------------
// Mamba3Block (DiT AdaLN-Zero style) for MI455X / gfx1250
//   - all big GEMMs on V_WMMA_F32_16X16X32_BF16 (wave32, f32 accum)
//   - one-time weight fp32->bf16 transpose (B staged like A)
//   - double-buffered LDS fed by GLOBAL_LOAD_ASYNC_TO_LDS_B128 (ASYNCcnt),
//     global_prefetch of k+2 tiles
//   - fused bias + GELU / residual-gate epilogues
// ---------------------------------------------------------------------------

#define USE_ASYNC_LDS 1   // 0 -> fall back to VGPR-staged double buffering

#define DIM     1024
#define NBATCH  4
#define SEQ     2048
#define NTOK    (NBATCH * SEQ)     // 8192 token rows
#define HDIM    2048               // FF hidden
#define SIXD    (6 * DIM)

typedef __attribute__((ext_vector_type(16))) __bf16 v16bf;
typedef __attribute__((ext_vector_type(8)))  __bf16 v8bf;
typedef __attribute__((ext_vector_type(8)))  float  v8f;

__device__ __forceinline__ float gelu_exact(float v) {
    return 0.5f * v * (1.0f + erff(v * 0.70710678118654752f));
}
__device__ __forceinline__ float gelu_tanh(float v) {
    float c = v + 0.044715f * v * v * v;
    return 0.5f * v * (1.0f + tanhf(0.7978845608028654f * c));
}
__device__ __forceinline__ v8bf ldg8(const __bf16* p) {
    return *reinterpret_cast<const v8bf*>(p);
}

#if USE_ASYNC_LDS
// Each lane copies 16 B global -> LDS without touching VGPRs; ASYNCcnt-tracked.
__device__ __forceinline__ void async_ld_b128(unsigned lds_off, const __bf16* gptr) {
    asm volatile("global_load_async_to_lds_b128 %0, %1, off"
                 :: "v"(lds_off), "v"((unsigned long long)(size_t)gptr)
                 : "memory");
}
__device__ __forceinline__ void wait_async0() {
    asm volatile("s_wait_asynccnt 0x0" ::: "memory");
}
#endif

// ---------------------------------------------------------------------------
// wt[n][k] = (bf16) w[k][n]   -- 32x32 LDS tile transpose, both sides coalesced
// ---------------------------------------------------------------------------
__global__ __launch_bounds__(256) void wconv_kernel(
    const float* __restrict__ w, __bf16* __restrict__ wt, int K, int N)
{
    __shared__ __bf16 tile[32][33];
    const int k0 = blockIdx.y * 32;
    const int n0 = blockIdx.x * 32;
    const int tr = threadIdx.x >> 5;    // 0..7
    const int tc = threadIdx.x & 31;    // 0..31
    #pragma unroll
    for (int r = 0; r < 32; r += 8)
        tile[tr + r][tc] = (__bf16)w[(size_t)(k0 + tr + r) * N + n0 + tc];
    __syncthreads();
    #pragma unroll
    for (int r = 0; r < 32; r += 8)
        wt[(size_t)(n0 + tr + r) * K + k0 + tc] = tile[tc][tr + r];
}

// ---------------------------------------------------------------------------
// emb[b, 6d] = SiLU(t[b,:]) @ ada_w + ada_b       (tiny: 4 x 6144)
// ---------------------------------------------------------------------------
__global__ __launch_bounds__(256) void ada_kernel(
    const float* __restrict__ t, const float* __restrict__ w,
    const float* __restrict__ bias, float* __restrict__ emb)
{
    __shared__ float st[DIM];
    const int blocks_per_b = SIXD / 256;            // 24
    const int b = blockIdx.x / blocks_per_b;
    const int j = (blockIdx.x % blocks_per_b) * 256 + threadIdx.x;

    for (int i = threadIdx.x; i < DIM; i += 256) {
        float tv = t[b * DIM + i];
        st[i] = tv / (1.0f + __expf(-tv));          // SiLU
    }
    __syncthreads();

    float acc = bias[j];
    #pragma unroll 4
    for (int k = 0; k < DIM; ++k)
        acc = fmaf(st[k], w[k * SIXD + j], acc);
    emb[b * SIXD + j] = acc;
}

// ---------------------------------------------------------------------------
// out[row,:] = LN(x[row,:]) * (1 + scale[b,:]) + shift[b,:]   -> bf16
// ---------------------------------------------------------------------------
__global__ __launch_bounds__(256) void ln_mod_kernel(
    const float* __restrict__ x, const float* __restrict__ emb,
    int shift_off, int scale_off, __bf16* __restrict__ out)
{
    const int row = blockIdx.x;
    const int b   = row >> 11;                      // SEQ = 2048 rows / batch
    const float* xr = x + (size_t)row * DIM;
    __shared__ float red[256];

    float v[4];
    float s = 0.0f, s2 = 0.0f;
    #pragma unroll
    for (int i = 0; i < 4; ++i) {
        v[i] = xr[threadIdx.x + i * 256];
        s  += v[i];
        s2 += v[i] * v[i];
    }
    red[threadIdx.x] = s;
    __syncthreads();
    for (int o = 128; o > 0; o >>= 1) {
        if (threadIdx.x < o) red[threadIdx.x] += red[threadIdx.x + o];
        __syncthreads();
    }
    const float mu = red[0] * (1.0f / DIM);
    __syncthreads();
    red[threadIdx.x] = s2;
    __syncthreads();
    for (int o = 128; o > 0; o >>= 1) {
        if (threadIdx.x < o) red[threadIdx.x] += red[threadIdx.x + o];
        __syncthreads();
    }
    const float var = red[0] * (1.0f / DIM) - mu * mu;
    const float rs  = rsqrtf(var + 1e-6f);

    const float* sh = emb + b * SIXD + shift_off;
    const float* sc = emb + b * SIXD + scale_off;
    #pragma unroll
    for (int i = 0; i < 4; ++i) {
        int c = threadIdx.x + i * 256;
        float nv = (v[i] - mu) * rs * (1.0f + sc[c]) + sh[c];
        out[(size_t)row * DIM + c] = (__bf16)nv;
    }
}

// ---------------------------------------------------------------------------
// WMMA GEMM:  C[M,N] = A(bf16)[M,K] @ Wt(bf16)[N,K]^T  + bias, fused epilogue
//   256 threads = 8 wave32s, block tile 128x128, BK=32, double-buffered LDS
//   wave grid 4(M) x 2(N); each wave: 2x4 WMMA tiles (32x64 patch) = 8 WMMA/step
//   MODE 0: bf16 store of gelu_exact(val)       (ssm hidden)
//   MODE 1: bf16 store of gelu_tanh(val)        (ff hidden)
//   MODE 2: f32 store of res + gate * val       (residual adds)
// ---------------------------------------------------------------------------
#define BM 128
#define BN 128
#define BK 32
#define LDS_S 40   // padded stride (multiple of 8 keeps 16B alignment)

template <int MODE>
__global__ __launch_bounds__(256) void gemm_wmma(
    const __bf16* __restrict__ A, const __bf16* __restrict__ Wt,
    const float* __restrict__ bias, int N, int K,
    __bf16* __restrict__ out_bf, float* __restrict__ out_f,
    const float* __restrict__ res, const float* __restrict__ emb, int gate_off)
{
    __shared__ __bf16 sA[2][BM * LDS_S];   // [buf][row][k]
    __shared__ __bf16 sB[2][BN * LDS_S];   // [buf][n][k]

    const int tid   = threadIdx.x;
    const int lane  = tid & 31;
    const int wave  = tid >> 5;
    const int wm    = wave & 3;         // 0..3 -> M sub-tile (32 rows)
    const int wn    = wave >> 2;        // 0..1 -> N sub-tile (64 cols)
    const int lh    = lane & 15;        // row/col within 16
    const int khalf = lane >> 4;        // K-interleave half

    const int bm0 = blockIdx.y * BM;
    const int bn0 = blockIdx.x * BN;

    // staging: each thread moves 16 A elems + 16 B elems per K-step
    const int sr = tid >> 1;            // 0..127 (row of A / row of Wt)
    const int sc = (tid & 1) * 16;      // 0 or 16
    const __bf16* gA = A  + (size_t)(bm0 + sr) * K + sc;
    const __bf16* gB = Wt + (size_t)(bn0 + sr) * K + sc;
    const int sidx = sr * LDS_S + sc;

#if USE_ASYNC_LDS
    // LDS byte offsets for the async DMA destinations (generic ptr low 32 bits)
    const unsigned la0 = (unsigned)(size_t)&sA[0][sidx];
    const unsigned la1 = (unsigned)(size_t)&sA[1][sidx];
    const unsigned lb0 = (unsigned)(size_t)&sB[0][sidx];
    const unsigned lb1 = (unsigned)(size_t)&sB[1][sidx];
#endif

    const v8f vzero = {0.f, 0.f, 0.f, 0.f, 0.f, 0.f, 0.f, 0.f};
    v8f acc[2][4];
    #pragma unroll
    for (int mt = 0; mt < 2; ++mt)
        #pragma unroll
        for (int nt = 0; nt < 4; ++nt) acc[mt][nt] = vzero;

    // ---- prologue: stage K-tile 0 into buffer 0
#if USE_ASYNC_LDS
    async_ld_b128(la0,      gA);
    async_ld_b128(la0 + 16, gA + 8);
    async_ld_b128(lb0,      gB);
    async_ld_b128(lb0 + 16, gB + 8);
    wait_async0();
#else
    {
        v8bf a0 = ldg8(gA), a1 = ldg8(gA + 8);
        v8bf b0 = ldg8(gB), b1 = ldg8(gB + 8);
        *reinterpret_cast<v8bf*>(&sA[0][sidx])     = a0;
        *reinterpret_cast<v8bf*>(&sA[0][sidx + 8]) = a1;
        *reinterpret_cast<v8bf*>(&sB[0][sidx])     = b0;
        *reinterpret_cast<v8bf*>(&sB[0][sidx + 8]) = b1;
    }
#endif
    __syncthreads();

    int cur = 0;
    for (int k0 = BK; k0 <= K; k0 += BK) {
        const bool have_next = (k0 < K);
#if USE_ASYNC_LDS
        if (have_next) {  // DMA K-tile k0 into the other buffer, overlaps WMMA
            const unsigned lda = (cur == 0) ? la1 : la0;
            const unsigned ldb = (cur == 0) ? lb1 : lb0;
            async_ld_b128(lda,      gA + k0);
            async_ld_b128(lda + 16, gA + k0 + 8);
            async_ld_b128(ldb,      gB + k0);
            async_ld_b128(ldb + 16, gB + k0 + 8);
            if (k0 + BK < K) {  // prefetch tile k+2 toward L2
                __builtin_prefetch(gA + k0 + BK, 0, 3);
                __builtin_prefetch(gB + k0 + BK, 0, 3);
            }
        }
#else
        v8bf a0, a1, b0, b1;
        if (have_next) {
            a0 = ldg8(gA + k0); a1 = ldg8(gA + k0 + 8);
            b0 = ldg8(gB + k0); b1 = ldg8(gB + k0 + 8);
            if (k0 + BK < K) {
                __builtin_prefetch(gA + k0 + BK, 0, 3);
                __builtin_prefetch(gB + k0 + BK, 0, 3);
            }
        }
#endif

        // ---- fragments from current buffer (contiguous 16B LDS reads;
        //      lanes 0-15: K 0-7/16-23, lanes 16-31: K 8-15/24-31)
        v16bf afrag[2], bfrag[4];
        #pragma unroll
        for (int mt = 0; mt < 2; ++mt) {
            const int rrow = (wm * 32 + mt * 16 + lh) * LDS_S + khalf * 8;
            v8bf lo = *reinterpret_cast<const v8bf*>(&sA[cur][rrow]);
            v8bf hi = *reinterpret_cast<const v8bf*>(&sA[cur][rrow + 16]);
            #pragma unroll
            for (int i = 0; i < 8; ++i) { afrag[mt][i] = lo[i]; afrag[mt][i + 8] = hi[i]; }
        }
        #pragma unroll
        for (int nt = 0; nt < 4; ++nt) {
            const int ccol = (wn * 64 + nt * 16 + lh) * LDS_S + khalf * 8;
            v8bf lo = *reinterpret_cast<const v8bf*>(&sB[cur][ccol]);
            v8bf hi = *reinterpret_cast<const v8bf*>(&sB[cur][ccol + 16]);
            #pragma unroll
            for (int i = 0; i < 8; ++i) { bfrag[nt][i] = lo[i]; bfrag[nt][i + 8] = hi[i]; }
        }

        // ---- 8 WMMAs with A/B fragment reuse
        #pragma unroll
        for (int mt = 0; mt < 2; ++mt)
            #pragma unroll
            for (int nt = 0; nt < 4; ++nt)
                acc[mt][nt] = __builtin_amdgcn_wmma_f32_16x16x32_bf16(
                    false, afrag[mt], false, bfrag[nt],
                    (short)0, acc[mt][nt], false, false);

        if (have_next) {
#if USE_ASYNC_LDS
            wait_async0();      // own DMA landed
            __syncthreads();    // everyone's DMA landed
#else
            const int nxt = cur ^ 1;
            *reinterpret_cast<v8bf*>(&sA[nxt][sidx])     = a0;
            *reinterpret_cast<v8bf*>(&sA[nxt][sidx + 8]) = a1;
            *reinterpret_cast<v8bf*>(&sB[nxt][sidx])     = b0;
            *reinterpret_cast<v8bf*>(&sB[nxt][sidx + 8]) = b1;
            __syncthreads();
#endif
            cur ^= 1;
        }
    }

    // ---- fused epilogue (C layout: VGPR r -> M = r (lanes<16) / r+8, N = lh)
    #pragma unroll
    for (int mt = 0; mt < 2; ++mt) {
        #pragma unroll
        for (int nt = 0; nt < 4; ++nt) {
            #pragma unroll
            for (int r = 0; r < 8; ++r) {
                int row = bm0 + wm * 32 + mt * 16 + ((lane < 16) ? r : r + 8);
                int col = bn0 + wn * 64 + nt * 16 + lh;
                float val = acc[mt][nt][r] + bias[col];
                if (MODE == 0) {
                    out_bf[(size_t)row * N + col] = (__bf16)gelu_exact(val);
                } else if (MODE == 1) {
                    out_bf[(size_t)row * N + col] = (__bf16)gelu_tanh(val);
                } else {
                    int bt = row >> 11;
                    float g = emb[bt * SIXD + gate_off + col];
                    out_f[(size_t)row * N + col] =
                        res[(size_t)row * N + col] + g * val;
                }
            }
        }
    }
}

// ---------------------------------------------------------------------------
// launch
// ---------------------------------------------------------------------------
extern "C" void kernel_launch(void* const* d_in, const int* in_sizes, int n_in,
                              void* d_out, int out_size, void* d_ws, size_t ws_size,
                              hipStream_t stream)
{
    const float* x      = (const float*)d_in[0];
    const float* t      = (const float*)d_in[1];
    const float* ada_w  = (const float*)d_in[2];
    const float* ada_b  = (const float*)d_in[3];
    const float* ssm_w1 = (const float*)d_in[4];
    const float* ssm_b1 = (const float*)d_in[5];
    const float* ssm_w2 = (const float*)d_in[6];
    const float* ssm_b2 = (const float*)d_in[7];
    const float* ff_w1  = (const float*)d_in[8];
    const float* ff_b1  = (const float*)d_in[9];
    const float* ff_w2  = (const float*)d_in[10];
    const float* ff_b2  = (const float*)d_in[11];
    float* out = (float*)d_out;

    char* ws = (char*)d_ws;
    float*  emb     = (float*) (ws);                         //   96 KB
    float*  x2      = (float*) (ws + (size_t)(1u   << 20));  //   32 MB fp32
    __bf16* normbuf = (__bf16*)(ws + (size_t)(36u  << 20));  //   16 MB bf16
    __bf16* hbuf    = (__bf16*)(ws + (size_t)(56u  << 20));  //   32 MB bf16
    __bf16* w1t     = (__bf16*)(ws + (size_t)(92u  << 20));  //    2 MB [1024][1024]
    __bf16* w2t     = (__bf16*)(ws + (size_t)(94u  << 20));  //    2 MB [1024][1024]
    __bf16* f1t     = (__bf16*)(ws + (size_t)(96u  << 20));  //    4 MB [2048][1024]
    __bf16* f2t     = (__bf16*)(ws + (size_t)(100u << 20));  //    4 MB [1024][2048]

    // 0. one-time weight transpose+convert: wt[n][k] = bf16(w[k][n])
    wconv_kernel<<<dim3(DIM  / 32, DIM  / 32), 256, 0, stream>>>(ssm_w1, w1t, DIM,  DIM);
    wconv_kernel<<<dim3(DIM  / 32, DIM  / 32), 256, 0, stream>>>(ssm_w2, w2t, DIM,  DIM);
    wconv_kernel<<<dim3(HDIM / 32, DIM  / 32), 256, 0, stream>>>(ff_w1,  f1t, DIM,  HDIM);
    wconv_kernel<<<dim3(DIM  / 32, HDIM / 32), 256, 0, stream>>>(ff_w2,  f2t, HDIM, DIM);

    // 1. AdaLN embedding: emb = SiLU(t) @ ada_w + ada_b
    ada_kernel<<<NBATCH * SIXD / 256, 256, 0, stream>>>(t, ada_w, ada_b, emb);

    // 2. norm = LN(x)*(1+scale_msa)+shift_msa  (bf16)
    ln_mod_kernel<<<NTOK, 256, 0, stream>>>(x, emb, 0 * DIM, 1 * DIM, normbuf);

    // 3. h = gelu_exact(norm @ ssm_w1 + b1)    (bf16)
    gemm_wmma<0><<<dim3(DIM / BN, NTOK / BM), 256, 0, stream>>>(
        normbuf, w1t, ssm_b1, DIM, DIM, hbuf, nullptr, nullptr, nullptr, 0);

    // 4. x2 = x + gate_msa * (h @ ssm_w2 + b2) (fp32)
    gemm_wmma<2><<<dim3(DIM / BN, NTOK / BM), 256, 0, stream>>>(
        hbuf, w2t, ssm_b2, DIM, DIM, nullptr, x2, x, emb, 2 * DIM);

    // 5. norm2 = LN(x2)*(1+scale_mlp)+shift_mlp (bf16)
    ln_mod_kernel<<<NTOK, 256, 0, stream>>>(x2, emb, 3 * DIM, 4 * DIM, normbuf);

    // 6. hf = gelu_tanh(norm2 @ ff_w1 + b1)    (bf16, N = 2048)
    gemm_wmma<1><<<dim3(HDIM / BN, NTOK / BM), 256, 0, stream>>>(
        normbuf, f1t, ff_b1, HDIM, DIM, hbuf, nullptr, nullptr, nullptr, 0);

    // 7. out = x2 + gate_mlp * (hf @ ff_w2 + b2) (fp32, K = 2048)
    gemm_wmma<2><<<dim3(DIM / BN, NTOK / BM), 256, 0, stream>>>(
        hbuf, f2t, ff_b2, DIM, HDIM, nullptr, out, x2, emb, 5 * DIM);
}
